// CPULiquidLayer_50929722196035
// MI455X (gfx1250) — matrix operand order repaired and checked by reading, not compiled, not run on hardware
//
#include <hip/hip_runtime.h>
#include <cmath>

// ---------------------------------------------------------------------------
// CT-RNN / liquid cell for MI455X (gfx1250, wave32, WMMA).
//   B=32, S=1024, I=512, H=1024
// Phase 1: h_in_all = x @ W_in  -> bf16 WMMA GEMM, memory-bound (~10us HBM).
// Phase 2: serial scan over S=1024 -> single persistent kernel, 128 waves
//   (16 blocks), grid barrier per step. W_rec^T strip (128KB/block) is staged
//   into LDS ONCE (320KB/WGP available) so the per-step serial chain reads the
//   B operand from LDS (ds_load_b128) instead of L2; the A operand (h state,
//   rewritten each step by other WGPs) is software-pipelined from global.
// ---------------------------------------------------------------------------

typedef __attribute__((ext_vector_type(16))) __bf16 v16bf;
typedef __attribute__((ext_vector_type(8)))  float  v8f;

union FragBF { uint4 u[2]; v16bf v; };

static constexpr int   B_  = 32;
static constexpr int   S_  = 1024;
static constexpr int   I_  = 512;
static constexpr int   H_  = 1024;
static constexpr float DT_ = 0.05f;

__device__ __forceinline__ unsigned short f2bf(float f) {
  union { float f; unsigned u; } c; c.f = f;
  unsigned r = c.u + 0x7FFFu + ((c.u >> 16) & 1u);   // round-to-nearest-even
  return (unsigned short)(r >> 16);
}

// ---- init: zero both h-state buffers, compute alpha = exp(-DT/tau), reset barrier
__global__ void k_init(const float* __restrict__ tau, float* __restrict__ alpha,
                       float* __restrict__ h32, unsigned short* __restrict__ h16,
                       unsigned* __restrict__ counter) {
  int i = blockIdx.x * blockDim.x + threadIdx.x;
  if (i < 2 * B_ * H_) { h32[i] = 0.0f; h16[i] = 0; }
  if (i < H_)          alpha[i] = __expf(-DT_ / tau[i]);
  if (i == 0)          *counter = 0u;
}

// ---- f32 -> bf16 raw convert
__global__ void k_cvt(const float* __restrict__ src, unsigned short* __restrict__ dst, int n) {
  int i = blockIdx.x * blockDim.x + threadIdx.x;
  if (i < n) dst[i] = f2bf(src[i]);
}

// ---- transpose + convert: src [R][C] f32 -> dst [C][R] bf16
__global__ void k_transpose_cvt(const float* __restrict__ src, unsigned short* __restrict__ dst,
                                int R, int C) {
  int i = blockIdx.x * blockDim.x + threadIdx.x;
  if (i >= R * C) return;
  int r = i / C, c = i % C;
  dst[(size_t)c * R + r] = f2bf(src[i]);
}

// ---- Phase 1: out[BS][H] = xbf[BS][I] * Win  (WinT stored [H][I], WinT[n][k]=Win[k][n])
// One wave computes a 16x64 tile: 1 A-frag reused across 4 B-frags per K-step.
__global__ void __launch_bounds__(256) k_gemm_in(
    const unsigned short* __restrict__ xbf,
    const unsigned short* __restrict__ WinT,
    float* __restrict__ out)
{
  const int lane = threadIdx.x & 31;
  const int half = lane >> 4;
  const int lm   = lane & 15;
  const int wid  = blockIdx.x * 8 + (threadIdx.x >> 5);
  const int m0   = (wid >> 4) * 16;       // 2048 m-tiles
  const int n0   = (wid & 15) * 64;       // 16 n-strips of 64

  const unsigned short* arow = xbf  + (size_t)(m0 + lm) * I_;
  const unsigned short* br0  = WinT + (size_t)(n0 +  0 + lm) * I_;
  const unsigned short* br1  = WinT + (size_t)(n0 + 16 + lm) * I_;
  const unsigned short* br2  = WinT + (size_t)(n0 + 32 + lm) * I_;
  const unsigned short* br3  = WinT + (size_t)(n0 + 48 + lm) * I_;

  const int aoff = half * 8;    // A frag: K +0/+8 halves, second b128 at +16
  const int boff = half * 8;    // B frag: K +0/+8 halves, second b128 at +16

  v8f c0 = {}, c1 = {}, c2 = {}, c3 = {};

  for (int kb = 0; kb < I_; kb += 32) {
    __builtin_prefetch((const void*)(arow + kb + 128), 0, 1);
    FragBF a, b0, b1, b2, b3;
    a.u[0]  = *reinterpret_cast<const uint4*>(arow + kb + aoff);
    a.u[1]  = *reinterpret_cast<const uint4*>(arow + kb + 16 + aoff);
    b0.u[0] = *reinterpret_cast<const uint4*>(br0 + kb + boff);
    b0.u[1] = *reinterpret_cast<const uint4*>(br0 + kb + boff + 16);
    b1.u[0] = *reinterpret_cast<const uint4*>(br1 + kb + boff);
    b1.u[1] = *reinterpret_cast<const uint4*>(br1 + kb + boff + 16);
    b2.u[0] = *reinterpret_cast<const uint4*>(br2 + kb + boff);
    b2.u[1] = *reinterpret_cast<const uint4*>(br2 + kb + boff + 16);
    b3.u[0] = *reinterpret_cast<const uint4*>(br3 + kb + boff);
    b3.u[1] = *reinterpret_cast<const uint4*>(br3 + kb + boff + 16);
    c0 = __builtin_amdgcn_wmma_f32_16x16x32_bf16(false, a.v, false, b0.v, (short)0, c0, false, false);
    c1 = __builtin_amdgcn_wmma_f32_16x16x32_bf16(false, a.v, false, b1.v, (short)0, c1, false, false);
    c2 = __builtin_amdgcn_wmma_f32_16x16x32_bf16(false, a.v, false, b2.v, (short)0, c2, false, false);
    c3 = __builtin_amdgcn_wmma_f32_16x16x32_bf16(false, a.v, false, b3.v, (short)0, c3, false, false);
  }

#pragma unroll
  for (int v = 0; v < 8; ++v) {
    const size_t row = (size_t)(m0 + half * 8 + v) * H_;
    out[row + n0 +  0 + lm] = c0[v];
    out[row + n0 + 16 + lm] = c1[v];
    out[row + n0 + 32 + lm] = c2[v];
    out[row + n0 + 48 + lm] = c3[v];
  }
}

// ---- Phase 2: persistent scan. 16 blocks x 8 waves = 128 waves = 2x64 WMMA tiles
// per step. Block's 64 WrecT rows (128KB) staged to LDS once; B operand from LDS.
// A operand (double-buffered h state) software-pipelined from global.
__global__ void __launch_bounds__(256) k_scan(
    const unsigned short* __restrict__ WrecT,   // [H][H], WrecT[n][k] = W_rec[k][n]
    const float* __restrict__ bias,
    const float* __restrict__ alpha,
    float* __restrict__ out,                    // [B][S][H] (holds h_in on entry)
    float* __restrict__ hlast,                  // [B][H]
    float* __restrict__ h32,                    // 2 x [B][H]
    unsigned short* __restrict__ h16,           // 2 x [B][H]
    unsigned* __restrict__ counter)
{
  extern __shared__ unsigned short lds_b[];     // [64][H] = 128 KB (of 320 KB/WGP)

  const int lane = threadIdx.x & 31;
  const int half = lane >> 4;
  const int lm   = lane & 15;
  const int wave = threadIdx.x >> 5;            // 0..7
  const int wid  = blockIdx.x * 8 + wave;       // 0..127
  const int m0   = (wid & 1) * 16;              // batch tile
  const int n0   = (wid >> 1) * 16;             // hidden tile (global)
  const int n    = n0 + lm;

  // ---- one-time stage: 64 contiguous WrecT rows for this block -> LDS
  {
    const uint4* src = reinterpret_cast<const uint4*>(WrecT + (size_t)(blockIdx.x * 64) * H_);
    uint4*       dst = reinterpret_cast<uint4*>(lds_b);
    for (int i = threadIdx.x; i < (64 * H_) / 8; i += blockDim.x) dst[i] = src[i];
  }
  __syncthreads();

  // this wave's B strip inside LDS: local n-strip = wave>>1, row = lm
  const unsigned short* lb = lds_b + (size_t)((wave >> 1) * 16 + lm) * H_;

  const float al = alpha[n];
  const float bi = bias[n];
  const int aoff = half * 8;    // A frag: K +0/+8 halves, second b128 at +16
  const int boff = half * 8;    // B frag: K +0/+8 halves, second b128 at +16
  const unsigned nb = gridDim.x;

  for (int t = 0; t < S_; ++t) {
    const int cur = t & 1, nxt = cur ^ 1;
    const unsigned short* hb  = h16 + (size_t)cur * (B_ * H_);
    unsigned short*       hbn = h16 + (size_t)nxt * (B_ * H_);
    const float*          hf  = h32 + (size_t)cur * (B_ * H_);
    float*                hfn = h32 + (size_t)nxt * (B_ * H_);

    const unsigned short* arow = hb + (size_t)(m0 + lm) * H_;
    v8f acc = {};

    // software-pipelined K loop: global A for k+32 overlaps WMMA(k) + LDS B(k)
    FragBF acur, anxt;
    acur.u[0] = *reinterpret_cast<const uint4*>(arow + aoff);
    acur.u[1] = *reinterpret_cast<const uint4*>(arow + 16 + aoff);
    for (int kb = 0; kb < H_; kb += 32) {
      const int kn = kb + 32;
      if (kn < H_) {
        anxt.u[0] = *reinterpret_cast<const uint4*>(arow + kn + aoff);
        anxt.u[1] = *reinterpret_cast<const uint4*>(arow + kn + 16 + aoff);
      }
      FragBF b;
      b.u[0] = *reinterpret_cast<const uint4*>(lb + kb + boff);
      b.u[1] = *reinterpret_cast<const uint4*>(lb + kb + boff + 16);
      acc = __builtin_amdgcn_wmma_f32_16x16x32_bf16(false, acur.v, false, b.v,
                                                    (short)0, acc, false, false);
      acur = anxt;
    }

#pragma unroll
    for (int v = 0; v < 8; ++v) {
      const int m = m0 + half * 8 + v;
      const size_t sidx = (size_t)m * H_ + n;
      const size_t oidx = ((size_t)m * S_ + t) * H_ + n;
      const float htot = tanhf(out[oidx] + acc[v] + bi);   // h_in + h@W_rec + bias
      const float hnew = al * hf[sidx] + (1.0f - al) * htot;
      out[oidx] = hnew;
      hfn[sidx] = hnew;
      hbn[sidx] = f2bf(hnew);
      if (t == S_ - 1) hlast[sidx] = hnew;
    }

    // device-wide barrier (monotonic counter: target = nb*(t+1))
    __syncthreads();
    if (threadIdx.x == 0) {
      __threadfence();                       // release: publish h writes
      atomicAdd(counter, 1u);
      const unsigned target = nb * (unsigned)(t + 1);
      while (__hip_atomic_load(counter, __ATOMIC_RELAXED, __HIP_MEMORY_SCOPE_AGENT) < target) {
        __builtin_amdgcn_s_sleep(1);
      }
      __threadfence();                       // acquire: invalidate WGP$ before rereads
    }
    __syncthreads();
  }
}

extern "C" void kernel_launch(void* const* d_in, const int* in_sizes, int n_in,
                              void* d_out, int out_size, void* d_ws, size_t ws_size,
                              hipStream_t stream) {
  (void)in_sizes; (void)n_in; (void)out_size; (void)ws_size;
  const float* x     = (const float*)d_in[0];
  const float* W_in  = (const float*)d_in[1];
  const float* W_rec = (const float*)d_in[2];
  const float* bias  = (const float*)d_in[3];
  const float* tau   = (const float*)d_in[4];
  float* out   = (float*)d_out;
  float* hlast = out + (size_t)B_ * S_ * H_;

  // Workspace carve-up (~37.2 MB total)
  char* ws = (char*)d_ws;
  auto alignup = [](size_t v) { return (v + 255) & ~(size_t)255; };
  size_t off = 0;
  unsigned short* xbf   = (unsigned short*)(ws + off); off = alignup(off + (size_t)2 * B_ * S_ * I_);
  unsigned short* WinT  = (unsigned short*)(ws + off); off = alignup(off + (size_t)2 * H_ * I_);
  unsigned short* WrecT = (unsigned short*)(ws + off); off = alignup(off + (size_t)2 * H_ * H_);
  float*          alpha = (float*)(ws + off);          off = alignup(off + (size_t)4 * H_);
  float*          h32   = (float*)(ws + off);          off = alignup(off + (size_t)4 * 2 * B_ * H_);
  unsigned short* h16   = (unsigned short*)(ws + off); off = alignup(off + (size_t)2 * 2 * B_ * H_);
  unsigned*       cnt   = (unsigned*)(ws + off);

  k_init<<<(2 * B_ * H_ + 255) / 256, 256, 0, stream>>>(tau, alpha, h32, h16, cnt);

  const int nx = B_ * S_ * I_;
  k_cvt<<<(nx + 255) / 256, 256, 0, stream>>>(x, xbf, nx);
  k_transpose_cvt<<<(I_ * H_ + 255) / 256, 256, 0, stream>>>(W_in,  WinT,  I_, H_);
  k_transpose_cvt<<<(H_ * H_ + 255) / 256, 256, 0, stream>>>(W_rec, WrecT, H_, H_);

  // 2048 m-tiles x 16 n-strips = 32768 waves, 8 waves/block -> 4096 blocks
  k_gemm_in<<<4096, 256, 0, stream>>>(xbf, WinT, out);

  // persistent scan: exactly 16 blocks, 128KB dynamic LDS each (WGP has 320KB)
  k_scan<<<16, 256, 64 * H_ * sizeof(unsigned short), stream>>>(
      WrecT, bias, alpha, out, hlast, h32, h16, cnt);
}
